// Evaluation_61933428408888
// MI455X (gfx1250) — matrix operand order repaired and checked by reading, not compiled, not run on hardware
//
#include <hip/hip_runtime.h>

typedef __attribute__((ext_vector_type(16))) _Float16 v16h;
typedef __attribute__((ext_vector_type(8)))  _Float16 v8h;
typedef __attribute__((ext_vector_type(8)))  float    v8f;

// ---------------------------------------------------------------- helpers

__device__ __forceinline__ v8f wmma_f16f32(v16h a, v16h b, v8f c) {
  // D = A(16x32 f16) * B(32x16 f16) + C(16x16 f32)
  return __builtin_amdgcn_wmma_f32_16x16x32_f16(
      false, a, false, b, (short)0, c, false, false);
}

__device__ __forceinline__ v8f splat8(float x) {
  v8f r;
#pragma unroll
  for (int i = 0; i < 8; ++i) r[i] = x;
  return r;
}

// A-matrix 16x32 f16 fragment (ISA 7.12.2): lane&15 = M row; kb = +8 for hi
// half-wave; halves[0..7] = K kb..kb+7, halves[8..15] = K kb+16..kb+23.
__device__ __forceinline__ v16h load_a_frag(const _Float16* base, int ld,
                                            int k0, int lane) {
  int m  = lane & 15;
  int kb = k0 + ((lane >> 4) << 3);
  const _Float16* p = base + m * ld + kb;
  v8h lo = *(const v8h*)p;
  v8h hi = *(const v8h*)(p + 16);
  v16h a;
#pragma unroll
  for (int i = 0; i < 8; ++i) { a[i] = lo[i]; a[i + 8] = hi[i]; }
  return a;
}

// B-matrix 32x16 f16 fragment from row-major W[N][K]: lane&15 = N col,
// kk = k0 (+8 for hi half-wave); halves[0..7] = K kk..kk+7,
// halves[8..15] = K kk+16..kk+23.  B[k][n] = W[n][k].
__device__ __forceinline__ v16h load_b_frag(const _Float16* W, int ld,
                                            int n0, int k0, int lane) {
  int n  = n0 + (lane & 15);
  int kk = k0 + ((lane >> 4) << 3);
  const _Float16* p = W + n * ld + kk;
  v8h lo = *(const v8h*)p;
  v8h hi = *(const v8h*)(p + 16);
  v16h b;
#pragma unroll
  for (int i = 0; i < 8; ++i) { b[i] = lo[i]; b[i + 8] = hi[i]; }
  return b;
}

__device__ __forceinline__ float sigmoidf_(float x) {
  return 1.f / (1.f + __expf(-x));
}

// CDNA5 async global->LDS copy (ASYNCcnt-tracked, ISA ch.10 / 15.18.3).
// lds_off = byte offset into LDS (low 32 bits of generic shared pointer).
__device__ __forceinline__ void async_copy_b128(unsigned lds_off,
                                                const void* gptr) {
  asm volatile("global_load_async_to_lds_b128 %0, %1, off"
               :: "v"(lds_off), "v"(gptr)
               : "memory");
}
__device__ __forceinline__ void wait_async0() {
  asm volatile("s_wait_asynccnt 0" ::: "memory");
}

// ================================================================ Kernel A
// GRU2_1 recurrence (1024 sequential steps) + dot attention + softmax
// + weighted sum.  One workgroup per 16-row batch tile; 8 waves; wave w owns
// gate columns [16w,16w+16) of each gate; weights f16-resident in LDS.
// x tiles are double-buffered fp32 via async global->LDS copies overlapped
// with the WMMA recurrence, then converted to f16 per step.
__global__ __launch_bounds__(256) void k_gru1_attn(
    const float* __restrict__ eff,   // [16,64,256,128]
    const float* __restrict__ Wih1,  // [384,128]
    const float* __restrict__ Whh1,  // [384,128]
    const float* __restrict__ bih1,  // [384]
    const float* __restrict__ bhh1,  // [384]
    float* __restrict__ probs_out,   // d_out+256: [64,256,16]
    _Float16* __restrict__ seqeff)   // ws: [64,256,128] f16
{
  extern __shared__ char smem[];
  _Float16* wih = (_Float16*)smem;            // 384*128
  _Float16* whh = wih + 384 * 128;            // 384*128
  _Float16* xs  = whh + 384 * 128;            // 16 rel * 16 b * 128
  _Float16* hs  = xs + 16 * 16 * 128;         // 16*128
  float* bi = (float*)(hs + 16 * 128);        // 384
  float* bh = bi + 384;                       // 384
  float* lg = bh + 384;                       // 16*16
  float* pr = lg + 256;                       // 16*16
  float* xpp = pr + 256;                      // 2 * 2048 fp32 ping-pong

  const int tid = threadIdx.x;
  const int b0  = blockIdx.x * 16;

  for (int i = tid; i < 384 * 128; i += 256) {
    wih[i] = (_Float16)Wih1[i];
    whh[i] = (_Float16)Whh1[i];
  }
  for (int i = tid; i < 384; i += 256) { bi[i] = bih1[i]; bh[i] = bhh1[i]; }
  for (int i = tid; i < 16 * 128; i += 256) hs[i] = (_Float16)0.f;

  const int wave = tid >> 5, lane = tid & 31;
  const int hiofs = (lane >= 16) ? 8 : 0;
  const int nR  = wave << 4;                 // within-gate column tile base
  const int col = nR + (lane & 15);          // this lane's h column

  const unsigned xpp_lds0 = (unsigned)(uintptr_t)xpp;            // buf 0
  const unsigned xpp_lds1 = (unsigned)(uintptr_t)(xpp + 2048);   // buf 1

  // kick off async copy of (r=0, t=0) x tile: 8 KB = 512 x 16B chunks
  {
    const char* g = (const char*)(eff + ((size_t)0 * 64 + 0) * 256 * 128
                                      + (size_t)b0 * 128);
    for (int c = tid; c < 512; c += 256)
      async_copy_b128(xpp_lds0 + (unsigned)c * 16u, g + (size_t)c * 16);
  }
  __syncthreads();

  for (int t = 0; t < 64; ++t) {
    for (int r = 0; r < 16; ++r) {
      // wait for async tile (r,t) then convert fp32 -> f16 into xs[r]
      wait_async0();
      __syncthreads();
      const float* xb = xpp + ((r & 1) ? 2048 : 0);
      _Float16* xd = xs + (((size_t)(r * 16)) << 7);
      for (int i = tid; i < 2048; i += 256) xd[i] = (_Float16)xb[i];
      __syncthreads();

      // issue async copy of the next tile into the other buffer
      {
        int rn = r + 1, tn = t;
        if (rn == 16) { rn = 0; tn = t + 1; }
        if (tn < 64) {
          unsigned lb = (rn & 1) ? xpp_lds1 : xpp_lds0;
          const char* g = (const char*)(
              eff + (((size_t)rn * 64 + tn) * 256 + b0) * 128);
          for (int c = tid; c < 512; c += 256)
            async_copy_b128(lb + (unsigned)c * 16u, g + (size_t)c * 16);
        }
      }

      const _Float16* xr = xs + (((size_t)(r * 16)) << 7);
      v16h hA[4], xA[4];
#pragma unroll
      for (int c = 0; c < 4; ++c) {
        hA[c] = load_a_frag(hs, 128, 32 * c, lane);
        xA[c] = load_a_frag(xr, 128, 32 * c, lane);
      }
      v8f accR  = splat8(bi[col] + bh[col]);
      v8f accZ  = splat8(bi[128 + col] + bh[128 + col]);
      v8f accIN = splat8(bi[256 + col]);
      v8f accHN = splat8(bh[256 + col]);
#pragma unroll
      for (int c = 0; c < 4; ++c) {
        accR  = wmma_f16f32(hA[c], load_b_frag(whh, 128, nR,       32 * c, lane), accR);
        accR  = wmma_f16f32(xA[c], load_b_frag(wih, 128, nR,       32 * c, lane), accR);
        accZ  = wmma_f16f32(hA[c], load_b_frag(whh, 128, 128 + nR, 32 * c, lane), accZ);
        accZ  = wmma_f16f32(xA[c], load_b_frag(wih, 128, 128 + nR, 32 * c, lane), accZ);
        accIN = wmma_f16f32(xA[c], load_b_frag(wih, 128, 256 + nR, 32 * c, lane), accIN);
        accHN = wmma_f16f32(hA[c], load_b_frag(whh, 128, 256 + nR, 32 * c, lane), accHN);
      }
      float hold[8];
#pragma unroll
      for (int v = 0; v < 8; ++v)
        hold[v] = (float)hs[((v + hiofs) << 7) + col];
      __syncthreads();  // all reads of hs done before overwrite
#pragma unroll
      for (int v = 0; v < 8; ++v) {
        float rr = sigmoidf_(accR[v]);
        float zz = sigmoidf_(accZ[v]);
        float nn = tanhf(accIN[v] + rr * accHN[v]);
        hs[((v + hiofs) << 7) + col] = (_Float16)((1.f - zz) * nn + zz * hold[v]);
      }
      __syncthreads();
    }

    // attention: logits[b][r] = 10*tanh(dot(x[b][r], h2[b]))
    {
      int bl = tid >> 3;
      for (int rr = (tid & 7); rr < 16; rr += 8) {
        const _Float16* xp = xs + (((size_t)(rr * 16 + bl)) << 7);
        const _Float16* hp = hs + ((size_t)bl << 7);
        float s = 0.f;
        for (int e = 0; e < 128; ++e) s += (float)xp[e] * (float)hp[e];
        lg[bl * 16 + rr] = 10.f * tanhf(s);
      }
    }
    __syncthreads();
    if (tid < 16) {
      float mx = -1e30f;
      for (int rr = 0; rr < 16; ++rr) mx = fmaxf(mx, lg[tid * 16 + rr]);
      float ev[16], sum = 0.f;
      for (int rr = 0; rr < 16; ++rr) { ev[rr] = __expf(lg[tid * 16 + rr] - mx); sum += ev[rr]; }
      float inv = 1.f / sum;
      for (int rr = 0; rr < 16; ++rr) {
        float p = ev[rr] * inv;
        pr[tid * 16 + rr] = p;
        probs_out[(((size_t)t * 256) + b0 + tid) * 16 + rr] = p;
      }
    }
    __syncthreads();
    // seq_eff[b][e] = sum_r p[b][r] * x[b][r][e]
    {
      int bl = tid >> 3, e0 = (tid & 7) << 4;
      for (int e = e0; e < e0 + 16; ++e) {
        float s = 0.f;
        for (int rr = 0; rr < 16; ++rr)
          s += pr[bl * 16 + rr] * (float)xs[(((size_t)(rr * 16 + bl)) << 7) + e];
        seqeff[(((size_t)t * 256) + b0 + bl) * 128 + e] = (_Float16)s;
      }
    }
    __syncthreads();
  }
}

// ================================================================ Kernel B1
// gi2 = seq_eff @ W_ih2^T + b_ih2   (M=16384, K=128, N=60 padded to 64)
__global__ __launch_bounds__(128) void k_gi2_gemm(
    const _Float16* __restrict__ seqeff,  // [16384,128] f16
    const float* __restrict__ Wih2,       // [60,128]
    const float* __restrict__ bih2,       // [60]
    float* __restrict__ gi2)              // ws: [16384,64] f32
{
  extern __shared__ char smem[];
  _Float16* w = (_Float16*)smem;          // [64][128], rows 60..63 zero
  float* bs = (float*)(w + 64 * 128);     // [64]
  const int tid = threadIdx.x;
  for (int i = tid; i < 64 * 128; i += 128)
    w[i] = ((i >> 7) < 60) ? (_Float16)Wih2[i] : (_Float16)0.f;
  for (int i = tid; i < 64; i += 128) bs[i] = (i < 60) ? bih2[i] : 0.f;
  __syncthreads();

  const int wave = tid >> 5, lane = tid & 31;
  const int hiofs = (lane >= 16) ? 8 : 0;
  const int m0 = blockIdx.x * 16;
  const int n0 = wave * 16;
  v8f acc = splat8(bs[n0 + (lane & 15)]);
#pragma unroll
  for (int c = 0; c < 4; ++c) {
    v16h a = load_a_frag(seqeff + (size_t)m0 * 128, 128, 32 * c, lane);
    v16h b = load_b_frag(w, 128, n0, 32 * c, lane);
    acc = wmma_f16f32(a, b, acc);
  }
  const int cn = n0 + (lane & 15);
#pragma unroll
  for (int v = 0; v < 8; ++v)
    gi2[(size_t)(m0 + v + hiofs) * 64 + cn] = acc[v];
}

// ================================================================ Kernel B2
// GRU2_2 (hid 20, gi precomputed) + GRU2_4 (hid 1) + temporal softmax.
// One thread per batch element.
__global__ __launch_bounds__(256) void k_small_chain(
    const float* __restrict__ gi2,   // [16384,64]
    const float* __restrict__ Whh2,  // [60,20]
    const float* __restrict__ bhh2,  // [60]
    const float* __restrict__ Wih4,  // [3,20]
    const float* __restrict__ Whh4,  // [3,1]
    const float* __restrict__ bih4,  // [3]
    const float* __restrict__ bhh4,  // [3]
    float* __restrict__ p2_out,      // d_out+262400: [64,256]
    float* __restrict__ out1f,       // ws: [64,256,20] f32
    _Float16* __restrict__ out1w)    // ws: [64,256,32] f16 (scaled, padded)
{
  __shared__ float whh[60 * 20], bh[60], wi4[60], wh4[3], bi4[3], bh4[3];
  const int tid = threadIdx.x;
  for (int i = tid; i < 1200; i += 256) whh[i] = Whh2[i];
  if (tid < 60) { bh[tid] = bhh2[tid]; wi4[tid] = Wih4[tid]; }
  if (tid < 3)  { wh4[tid] = Whh4[tid]; bi4[tid] = bih4[tid]; bh4[tid] = bhh4[tid]; }
  __syncthreads();

  const int b = tid;
  float h[20];
#pragma unroll
  for (int k = 0; k < 20; ++k) h[k] = 0.f;
  float h4 = 0.f;
  float out2[64];

  for (int t = 0; t < 64; ++t) {
    const float* gi = gi2 + ((size_t)t * 256 + b) * 64;
    float g[60];
    for (int gg = 0; gg < 60; ++gg) {
      float s = bh[gg];
#pragma unroll
      for (int k = 0; k < 20; ++k) s += h[k] * whh[gg * 20 + k];
      g[gg] = s;
    }
    float hn[20];
#pragma unroll
    for (int k = 0; k < 20; ++k) {
      float rr = sigmoidf_(gi[k] + g[k]);
      float zz = sigmoidf_(gi[20 + k] + g[20 + k]);
      float nn = tanhf(gi[40 + k] + rr * g[40 + k]);
      hn[k] = (1.f - zz) * nn + zz * h[k];
    }
#pragma unroll
    for (int k = 0; k < 20; ++k) {
      h[k] = hn[k];
      out1f[((size_t)t * 256 + b) * 20 + k] = h[k];
    }
    // GRU2_4 on out_1 (hidden size 1)
    float gi4[3];
#pragma unroll
    for (int gg = 0; gg < 3; ++gg) {
      float s = bi4[gg];
#pragma unroll
      for (int k = 0; k < 20; ++k) s += wi4[gg * 20 + k] * h[k];
      gi4[gg] = s;
    }
    float r4 = sigmoidf_(gi4[0] + h4 * wh4[0] + bh4[0]);
    float z4 = sigmoidf_(gi4[1] + h4 * wh4[1] + bh4[1]);
    float n4 = tanhf(gi4[2] + r4 * (h4 * wh4[2] + bh4[2]));
    h4 = (1.f - z4) * n4 + z4 * h4;
    out2[t] = h4;
  }
  // temporal softmax over seq, write probs_2 and scaled GRU3 inputs
  float mx = -1e30f;
  for (int t = 0; t < 64; ++t) mx = fmaxf(mx, out2[t]);
  float sum = 0.f;
  for (int t = 0; t < 64; ++t) { out2[t] = __expf(out2[t] - mx); sum += out2[t]; }
  float inv = 1.f / sum;
  for (int t = 0; t < 64; ++t) {
    float p = out2[t] * inv;
    p2_out[t * 256 + b] = p;
    const float* o1 = out1f + ((size_t)t * 256 + b) * 20;
    _Float16* dst = out1w + ((size_t)t * 256 + b) * 32;
    for (int k = 0; k < 20; ++k) dst[k] = (_Float16)(o1[k] * p);
    for (int k = 20; k < 32; ++k) dst[k] = (_Float16)0.f;
  }
}

// ================================================================ Kernel C
// GRU2_3 recurrence (64 steps, in 20->pad32, hid 128) + MLP head, WMMA.
__global__ __launch_bounds__(256) void k_gru3_head(
    const _Float16* __restrict__ out1w,  // [64,256,32] f16
    const float* __restrict__ Wih3,      // [384,20]
    const float* __restrict__ Whh3,      // [384,128]
    const float* __restrict__ bih3,      // [384]
    const float* __restrict__ bhh3,      // [384]
    const float* __restrict__ W1,        // [256,128]
    const float* __restrict__ b1,        // [256]
    const float* __restrict__ W2,        // [1,256]
    const float* __restrict__ b2,        // [1]
    float* __restrict__ outs)            // d_out: [256]
{
  extern __shared__ char smem[];
  _Float16* wih = (_Float16*)smem;         // [384][32] (K padded)
  _Float16* whh = wih + 384 * 32;          // [384][128]
  _Float16* w1h = whh + 384 * 128;         // [256][128]
  _Float16* ins = w1h + 256 * 128;         // [16][32]
  _Float16* hs  = ins + 16 * 32;           // [16][128]
  float* bi  = (float*)(hs + 16 * 128);    // 384
  float* bhv = bi + 384;                   // 384
  float* b1s = bhv + 384;                  // 256
  float* w2s = b1s + 256;                  // 256
  float* o3  = w2s + 256;                  // [16][256]
  float* red = o3 + 16 * 256;              // [16][16]

  const int tid = threadIdx.x;
  const int b0  = blockIdx.x * 16;
  for (int i = tid; i < 384 * 32; i += 256) {
    int k = i & 31;
    wih[i] = (k < 20) ? (_Float16)Wih3[(i >> 5) * 20 + k] : (_Float16)0.f;
  }
  for (int i = tid; i < 384 * 128; i += 256) whh[i] = (_Float16)Whh3[i];
  for (int i = tid; i < 256 * 128; i += 256) w1h[i] = (_Float16)W1[i];
  for (int i = tid; i < 384; i += 256) { bi[i] = bih3[i]; bhv[i] = bhh3[i]; }
  for (int i = tid; i < 256; i += 256) { b1s[i] = b1[i]; w2s[i] = W2[i]; }
  for (int i = tid; i < 16 * 128; i += 256) hs[i] = (_Float16)0.f;
  __syncthreads();

  const int wave = tid >> 5, lane = tid & 31;
  const int hiofs = (lane >= 16) ? 8 : 0;
  const int nR  = wave << 4;
  const int col = nR + (lane & 15);

  for (int t = 0; t < 64; ++t) {
    for (int i = tid; i < 512; i += 256)
      ins[i] = out1w[((size_t)t * 256 + b0) * 32 + i];
    __syncthreads();

    v16h iA = load_a_frag(ins, 32, 0, lane);
    v16h hA[4];
#pragma unroll
    for (int c = 0; c < 4; ++c) hA[c] = load_a_frag(hs, 128, 32 * c, lane);

    v8f accR  = splat8(bi[col] + bhv[col]);
    v8f accZ  = splat8(bi[128 + col] + bhv[128 + col]);
    v8f accIN = splat8(bi[256 + col]);
    v8f accHN = splat8(bhv[256 + col]);
    accR  = wmma_f16f32(iA, load_b_frag(wih, 32, nR,       0, lane), accR);
    accZ  = wmma_f16f32(iA, load_b_frag(wih, 32, 128 + nR, 0, lane), accZ);
    accIN = wmma_f16f32(iA, load_b_frag(wih, 32, 256 + nR, 0, lane), accIN);
#pragma unroll
    for (int c = 0; c < 4; ++c) {
      accR  = wmma_f16f32(hA[c], load_b_frag(whh, 128, nR,       32 * c, lane), accR);
      accZ  = wmma_f16f32(hA[c], load_b_frag(whh, 128, 128 + nR, 32 * c, lane), accZ);
      accHN = wmma_f16f32(hA[c], load_b_frag(whh, 128, 256 + nR, 32 * c, lane), accHN);
    }
    float hold[8];
#pragma unroll
    for (int v = 0; v < 8; ++v) hold[v] = (float)hs[((v + hiofs) << 7) + col];
    __syncthreads();
#pragma unroll
    for (int v = 0; v < 8; ++v) {
      float rr = sigmoidf_(accR[v]);
      float zz = sigmoidf_(accZ[v]);
      float nn = tanhf(accIN[v] + rr * accHN[v]);
      hs[((v + hiofs) << 7) + col] = (_Float16)((1.f - zz) * nn + zz * hold[v]);
    }
    __syncthreads();
  }

  // head: out3 = relu(h4 @ W1^T + b1)  -> o3[16][256]
  {
    v16h hA[4];
#pragma unroll
    for (int c = 0; c < 4; ++c) hA[c] = load_a_frag(hs, 128, 32 * c, lane);
#pragma unroll
    for (int tt = 0; tt < 2; ++tt) {
      int n0 = (wave * 2 + tt) * 16;
      v8f acc = splat8(b1s[n0 + (lane & 15)]);
#pragma unroll
      for (int c = 0; c < 4; ++c)
        acc = wmma_f16f32(hA[c], load_b_frag(w1h, 128, n0, 32 * c, lane), acc);
#pragma unroll
      for (int v = 0; v < 8; ++v)
        o3[(v + hiofs) * 256 + n0 + (lane & 15)] = fmaxf(acc[v], 0.f);
    }
  }
  __syncthreads();
  {
    int bl = tid >> 4, n0 = (tid & 15) << 4;
    float s = 0.f;
    for (int n = n0; n < n0 + 16; ++n) s += o3[bl * 256 + n] * w2s[n];
    red[bl * 16 + (tid & 15)] = s;
  }
  __syncthreads();
  if (tid < 16) {
    float s = b2[0];
    for (int j = 0; j < 16; ++j) s += red[tid * 16 + j];
    outs[b0 + tid] = s;
  }
}

// ================================================================ launch

extern "C" void kernel_launch(void* const* d_in, const int* in_sizes, int n_in,
                              void* d_out, int out_size, void* d_ws, size_t ws_size,
                              hipStream_t stream) {
  const float* eff   = (const float*)d_in[0];
  const float* Wih1  = (const float*)d_in[1];
  const float* Whh1  = (const float*)d_in[2];
  const float* bih1  = (const float*)d_in[3];
  const float* bhh1  = (const float*)d_in[4];
  const float* Wih2  = (const float*)d_in[5];
  const float* Whh2  = (const float*)d_in[6];
  const float* bih2  = (const float*)d_in[7];
  const float* bhh2  = (const float*)d_in[8];
  const float* Wih3  = (const float*)d_in[9];
  const float* Whh3  = (const float*)d_in[10];
  const float* bih3  = (const float*)d_in[11];
  const float* bhh3  = (const float*)d_in[12];
  const float* Wih4  = (const float*)d_in[13];
  const float* Whh4  = (const float*)d_in[14];
  const float* bih4  = (const float*)d_in[15];
  const float* bhh4  = (const float*)d_in[16];
  const float* W1    = (const float*)d_in[17];
  const float* b1    = (const float*)d_in[18];
  const float* W2    = (const float*)d_in[19];
  const float* b2    = (const float*)d_in[20];

  float* out = (float*)d_out;
  float* probs_out = out + 256;                  // logits_1 region [64,256,16]
  float* p2_out    = out + 256 + 64 * 256 * 16;  // probs_2 region [64,256]

  char* ws = (char*)d_ws;
  _Float16* seqeff = (_Float16*)ws;                       // 4 MiB
  float*    gi2    = (float*)(ws + 4194304);              // 4 MiB
  float*    out1f  = (float*)(ws + 8388608);              // 1.25 MiB
  _Float16* out1w  = (_Float16*)(ws + 9699328);           // 1 MiB

  const size_t SMEM_A = 2 * 384 * 128 * 2 + 16 * 16 * 128 * 2 + 16 * 128 * 2
                      + (384 + 384 + 256 + 256) * 4
                      + 2 * 2048 * 4;                         // ~287.7 KB
  const size_t SMEM_B1 = 64 * 128 * 2 + 64 * 4;               // ~16.6 KB
  const size_t SMEM_C  = 384 * 32 * 2 + 384 * 128 * 2 + 256 * 128 * 2
                       + 16 * 32 * 2 + 16 * 128 * 2
                       + (384 + 384 + 256 + 256 + 16 * 256 + 256) * 4; // ~216 KB

  (void)hipFuncSetAttribute((const void*)k_gru1_attn,
      hipFuncAttributeMaxDynamicSharedMemorySize, (int)SMEM_A);
  (void)hipFuncSetAttribute((const void*)k_gru3_head,
      hipFuncAttributeMaxDynamicSharedMemorySize, (int)SMEM_C);

  k_gru1_attn<<<16, 256, SMEM_A, stream>>>(eff, Wih1, Whh1, bih1, bhh1,
                                           probs_out, seqeff);
  k_gi2_gemm<<<1024, 128, SMEM_B1, stream>>>(seqeff, Wih2, bih2, gi2);
  k_small_chain<<<1, 256, 0, stream>>>(gi2, Whh2, bhh2, Wih4, Whh4, bih4, bhh4,
                                       p2_out, out1f, out1w);
  k_gru3_head<<<16, 256, SMEM_C, stream>>>(out1w, Wih3, Whh3, bih3, bhh3,
                                           W1, b1, W2, b2, out);
}
